// DGCNN_cls_61950608277552
// MI455X (gfx1250) — matrix operand order repaired and checked by reading, not compiled, not run on hardware
//
#include <hip/hip_runtime.h>
#include <float.h>

#define BATCH 8
#define NPTS  1024
#define KNN   20
#define BNRS  0.9999950000374997f   // 1/sqrt(1+1e-5)
#define ASTR  40                    // LDS row stride (u16): 80B, 16B-aligned, low bank conflict

#if defined(__has_builtin)
#  if __has_builtin(__builtin_amdgcn_tensor_load_to_lds) && __has_builtin(__builtin_amdgcn_s_wait_tensorcnt)
#    define HAVE_TDM 1
#  endif
#endif
#ifndef HAVE_TDM
#  define HAVE_TDM 0
#endif

typedef __attribute__((ext_vector_type(16))) __bf16          v16bf;
typedef __attribute__((ext_vector_type(16))) unsigned short  v16us;
typedef __attribute__((ext_vector_type(8)))  unsigned short  v8us;
typedef __attribute__((ext_vector_type(4)))  unsigned short  v4us;
typedef __attribute__((ext_vector_type(8)))  float           v8f;
typedef __attribute__((ext_vector_type(4)))  float           v4f;
typedef __attribute__((ext_vector_type(4)))  unsigned int    v4u;
typedef __attribute__((ext_vector_type(4)))  int             v4i;
typedef __attribute__((ext_vector_type(8)))  int             v8i;

union BF16V { v16us u; v8us h[2]; v16bf b; };

__device__ __forceinline__ unsigned short f2bf(float f) {
  unsigned u = __float_as_uint(f);
  unsigned r = u + 0x7fffu + ((u >> 16) & 1u);   // RNE
  return (unsigned short)(r >> 16);
}
__device__ __forceinline__ float bf2f(unsigned short h) {
  return __uint_as_float(((unsigned)h) << 16);
}
__device__ __forceinline__ float leakyf(float x) { return x >= 0.f ? x : 0.2f * x; }

__device__ __forceinline__ v4us pack4(v4f v) {
  v4us r;
  #pragma unroll
  for (int j = 0; j < 4; ++j) r[j] = f2bf(v[j]);
  return r;
}

// order-preserving float atomic max via integer atomics (max is order-independent -> deterministic)
__device__ __forceinline__ void atomicMaxF(float* addr, float val) {
  if (val >= 0.f) atomicMax((int*)addr, __float_as_int(val));
  else            atomicMin((unsigned int*)addr, __float_as_uint(val));
}

// ---------------------------------------------------------------------------
// TDM: stage a (rows x 32) u16 tile, row stride `strideElems`, into LDS with
// hardware padding of 4 DWORDs every 16 DWORDs -> LDS row stride = 40 u16 = ASTR.
// D# per CDNA5 ISA 8.3/8.4. Issue from ONE wave (TDM ignores EXEC).
// ---------------------------------------------------------------------------
__device__ __forceinline__ void tdm_stage_weights(const unsigned short* gsrc,
                                                  unsigned short* lds,
                                                  int rows, int strideElems) {
#if HAVE_TDM
  unsigned long long ga = (unsigned long long)(size_t)gsrc;
  v4u g0;
  g0[0] = 1u;                                   // count=1, user descriptor
  g0[1] = (unsigned)(size_t)(void*)lds;         // lds_addr (segment offset in low 32b)
  g0[2] = (unsigned)ga;                         // global_addr[31:0]
  g0[3] = (unsigned)((ga >> 32) & 0x1FFFFFFu) | (2u << 30);  // addr[56:32] | type=2
  v8i g1;
  g1[0] = (int)((1u << 16) | (1u << 20) | (3u << 22) | (3u << 25));
  //            data_size=2B  pad_enable  pad every 16 DW  pad 4 DW
  g1[1] = (int)(32u << 16);                     // tensor_dim0 = 32 (lo16)
  g1[2] = (int)((unsigned)rows << 16);          // tensor_dim0 hi=0 | tensor_dim1 lo16
  g1[3] = (int)(32u << 16);                     // tensor_dim1 hi=0 | tile_dim0 = 32
  g1[4] = rows;                                 // tile_dim1 = rows, tile_dim2 = 0
  g1[5] = strideElems;                          // tensor_dim0_stride (lo32)
  g1[6] = 0;
  g1[7] = 0;
#if __clang_major__ >= 23
  __builtin_amdgcn_tensor_load_to_lds(g0, g1, (v4i)0, (v4i)0, (v8i)0, 0);
#else
  __builtin_amdgcn_tensor_load_to_lds(g0, g1, (v4i)0, (v4i)0, 0);
#endif
#endif
}

// fallback / non-TDM cooperative copy with the same padded layout
__device__ __forceinline__ void copy_stage_weights(const unsigned short* gsrc,
                                                   unsigned short* lds,
                                                   int rows, int strideElems,
                                                   int tid, int nthr) {
  for (int s = tid; s < rows * 4; s += nthr) {   // 8-u16 chunks
    int r = s >> 2, q = s & 3;
    *(v8us*)(lds + r * ASTR + q * 8) = *(const v8us*)(gsrc + (size_t)r * strideElems + q * 8);
  }
}

// ---------------------------------------------------------------------------
// small prep / glue kernels
// ---------------------------------------------------------------------------
__global__ void k_transpose_x(const float* __restrict__ x, float* __restrict__ xt) {
  int t = blockIdx.x * blockDim.x + threadIdx.x;
  if (t >= BATCH * NPTS) return;
  int b = t / NPTS, n = t % NPTS;
  xt[t * 4 + 0] = x[((size_t)b * 3 + 0) * NPTS + n];
  xt[t * 4 + 1] = x[((size_t)b * 3 + 1) * NPTS + n];
  xt[t * 4 + 2] = x[((size_t)b * 3 + 2) * NPTS + n];
  xt[t * 4 + 3] = 0.f;
}

// Wb[o*Kpad + k] = bf16(W[o*Kreal + k]) — ORIGINAL orientation, zero-padded
__global__ void k_convert_w(const float* __restrict__ W, unsigned short* __restrict__ Wb,
                            int Oreal, int Kreal, int Opad, int Kpad) {
  int t = blockIdx.x * blockDim.x + threadIdx.x;
  if (t >= Opad * Kpad) return;
  int o = t / Kpad, k = t % Kpad;
  float v = (o < Oreal && k < Kreal) ? W[(size_t)o * Kreal + k] : 0.f;
  Wb[t] = f2bf(v);
}

__global__ void k_rownorm(const float* __restrict__ X, int ld, int C,
                          float* __restrict__ xx, int rows) {
  int t = blockIdx.x * blockDim.x + threadIdx.x;
  if (t >= rows) return;
  const v4f* Xv = (const v4f*)(X + (size_t)t * ld);
  float s = 0.f;
  for (int c4 = 0; c4 < C / 4; ++c4) {
    v4f v = Xv[c4];
    #pragma unroll
    for (int j = 0; j < 4; ++j) s += v[j] * v[j];
  }
  for (int c = (C / 4) * 4; c < C; ++c) { float v = X[(size_t)t * ld + c]; s += v * v; }
  xx[t] = s;
}

__global__ void k_build_nf1(const float* __restrict__ xcat, float* __restrict__ nf1) {
  int t = blockIdx.x * blockDim.x + threadIdx.x;  // over v4f elements
  if (t >= BATCH * NPTS * 64) return;
  int row = t >> 6, col4 = t & 63;
  ((v4f*)nf1)[t] = ((const v4f*)xcat)[(size_t)row * 32 + (col4 & 31)];
}

__global__ void k_fill(float* __restrict__ p, int n, float v) {
  int t = blockIdx.x * blockDim.x + threadIdx.x;
  if (t < n) p[t] = v;
}

__global__ void k_build_f0(const float* __restrict__ p1, const float* __restrict__ p2,
                           float* __restrict__ f0) {
  int t = blockIdx.x * blockDim.x + threadIdx.x;
  if (t >= 16 * 2048) return;
  int r = t >> 11, c = t & 2047;
  float v = 0.f;
  if (r < 8) v = (c < 1024) ? p1[r * 1024 + c] : p2[r * 1024 + (c - 1024)];
  f0[t] = v;
}

// ---------------------------------------------------------------------------
// kNN selection (matches jax.lax.top_k tie-breaking: strict >, lower idx first)
// ---------------------------------------------------------------------------
__device__ __forceinline__ void topk_insert(float d, int m, float* bv, int* bi) {
  if (d > bv[KNN - 1]) {
    int pos = KNN - 1;
    while (pos > 0 && d > bv[pos - 1]) { bv[pos] = bv[pos - 1]; bi[pos] = bi[pos - 1]; --pos; }
    bv[pos] = d; bi[pos] = m;
  }
}

__global__ void k_knn_xyz(const float* __restrict__ xt, const float* __restrict__ xx,
                          int* __restrict__ idx) {
  int t = blockIdx.x * blockDim.x + threadIdx.x;
  if (t >= BATCH * NPTS) return;
  int b = t / NPTS, n = t % NPTS;
  const v4f* xb = (const v4f*)(xt + (size_t)b * NPTS * 4);
  v4f xc = xb[n];
  float xn = xx[t];
  const v4f* xxv = (const v4f*)(xx + (size_t)b * NPTS);
  float bv[KNN]; int bi[KNN];
  #pragma unroll
  for (int i = 0; i < KNN; ++i) { bv[i] = -FLT_MAX; bi[i] = 0; }
  for (int m4 = 0; m4 < NPTS / 4; ++m4) {
    v4f xm2 = xxv[m4];
    #pragma unroll
    for (int j = 0; j < 4; ++j) {
      int m = m4 * 4 + j;
      v4f xm = xb[m];
      float dot = xc[0] * xm[0] + xc[1] * xm[1] + xc[2] * xm[2];
      float d = 2.f * dot - xn - xm2[j];
      topk_insert(d, m, bv, bi);
    }
  }
  for (int i = 0; i < KNN; ++i) idx[(size_t)t * KNN + i] = bi[i];
}

__global__ void k_topk_gram(const float* __restrict__ G, const float* __restrict__ xx,
                            int* __restrict__ idx) {
  int t = blockIdx.x * blockDim.x + threadIdx.x;
  if (t >= BATCH * NPTS) return;
  int b = t / NPTS;
  const v4f* row = (const v4f*)(G + (size_t)t * NPTS);
  const v4f* xxv = (const v4f*)(xx + (size_t)b * NPTS);
  float xn = xx[t];
  float bv[KNN]; int bi[KNN];
  #pragma unroll
  for (int i = 0; i < KNN; ++i) { bv[i] = -FLT_MAX; bi[i] = 0; }
  for (int m4 = 0; m4 < NPTS / 4; ++m4) {
    v4f g4 = row[m4];
    v4f x4 = xxv[m4];
    #pragma unroll
    for (int j = 0; j < 4; ++j) {
      float d = 2.f * g4[j] - xn - x4[j];
      topk_insert(d, m4 * 4 + j, bv, bi);
    }
  }
  for (int i = 0; i < KNN; ++i) idx[(size_t)t * KNN + i] = bi[i];
}

// ---------------------------------------------------------------------------
// Gram matrix G = X X^T per batch, hi/lo bf16 split for ~fp32 accuracy
// (top-k neighbor ordering is the precision-critical step).
// block 256 threads (8 waves), 128x128 output tile; wave w: mtile w, ntiles 0..7
// ---------------------------------------------------------------------------
__global__ void k_gram_split(const float* __restrict__ X, int ld, int ksteps,
                             float* __restrict__ G) {
  __shared__ unsigned short Ahi[128 * ASTR], Alo[128 * ASTR], Bhi[128 * ASTR], Blo[128 * ASTR];
  const int tid = threadIdx.x, lane = tid & 31, wave = tid >> 5;
  const int lrow = lane & 15, lg = lane >> 4;
  const int b = blockIdx.z;
  const int rBase = blockIdx.x * 128, cBase = blockIdx.y * 128;

  v8f acc[8];
  #pragma unroll
  for (int t = 0; t < 8; ++t)
    #pragma unroll
    for (int j = 0; j < 8; ++j) acc[t][j] = 0.f;

  for (int ks = 0; ks < ksteps; ++ks) {
    __syncthreads();
    for (int s = tid; s < 128 * 8; s += 256) {     // float4 chunks
      int r = s >> 3, q = s & 7;
      int c4 = (ks * 32 + q * 4) >> 2;
      v4f va = ((const v4f*)X)[((size_t)b * NPTS + rBase + r) * (ld >> 2) + c4];
      v4f vb = ((const v4f*)X)[((size_t)b * NPTS + cBase + r) * (ld >> 2) + c4];
      v4us ha = pack4(va), hb = pack4(vb);
      v4f la, lb;
      #pragma unroll
      for (int j = 0; j < 4; ++j) { la[j] = va[j] - bf2f(ha[j]); lb[j] = vb[j] - bf2f(hb[j]); }
      int off = r * ASTR + q * 4;
      *(v4us*)(Ahi + off) = ha;  *(v4us*)(Alo + off) = pack4(la);
      *(v4us*)(Bhi + off) = hb;  *(v4us*)(Blo + off) = pack4(lb);
    }
    __syncthreads();
    BF16V ahi, alo;
    #pragma unroll
    for (int e = 0; e < 16; ++e) {
      int off = (wave * 16 + lrow) * ASTR + (lg * 16 + e);
      ahi.u[e] = Ahi[off]; alo.u[e] = Alo[off];
    }
    #pragma unroll
    for (int nt = 0; nt < 8; ++nt) {
      BF16V bhi, blo;
      #pragma unroll
      for (int e = 0; e < 16; ++e) {
        int off = (nt * 16 + lrow) * ASTR + (lg * 16 + e);
        bhi.u[e] = Bhi[off]; blo.u[e] = Blo[off];
      }
      acc[nt] = __builtin_amdgcn_wmma_f32_16x16x32_bf16(false, ahi.b, false, bhi.b, (short)0, acc[nt], false, false);
      acc[nt] = __builtin_amdgcn_wmma_f32_16x16x32_bf16(false, ahi.b, false, blo.b, (short)0, acc[nt], false, false);
      acc[nt] = __builtin_amdgcn_wmma_f32_16x16x32_bf16(false, alo.b, false, bhi.b, (short)0, acc[nt], false, false);
    }
  }
  #pragma unroll
  for (int nt = 0; nt < 8; ++nt)
    #pragma unroll
    for (int v = 0; v < 8; ++v) {
      int row = rBase + wave * 16 + lg * 8 + v;
      int col = cBase + nt * 16 + lrow;
      G[((size_t)b * NPTS + row) * NPTS + col] = acc[nt][v];
    }
}

// ---------------------------------------------------------------------------
// Fused EdgeConv: gather graph feature -> WMMA vs W(bf16, OxK, TDM-staged) ->
// BN+leaky -> max over k.  4 points/block, M = 4*20 = 80 rows = 5 M-tiles.
// LDS union: [As | Bs] during k-loop, per-wave Y buffers during reduction.
// ---------------------------------------------------------------------------
template <int CIN, int CPAD, int O>
__global__ void k_edgeconv(const float* __restrict__ X, int ldX,
                           const int* __restrict__ idx,
                           const unsigned short* __restrict__ Wb,   // O x CPAD bf16
                           const float* __restrict__ gam, const float* __restrict__ bet,
                           float* __restrict__ out, int ldout) {
  constexpr int NWAVES = (O >= 256) ? 8 : 4;
  constexpr int NTPW = O / (16 * NWAVES);
  constexpr int KS = CPAD / 32;
  constexpr int AS_ELEMS = 80 * ASTR;
  constexpr int BS_ELEMS = O * ASTR;
  constexpr size_t SM1 = (size_t)(AS_ELEMS + BS_ELEMS) * 2;
  constexpr size_t SM2 = (size_t)NWAVES * 80 * 16 * 4;
  constexpr size_t SMEM = SM1 > SM2 ? SM1 : SM2;
  __shared__ __align__(16) char smem[SMEM];
  unsigned short* As = (unsigned short*)smem;
  unsigned short* Bs = As + AS_ELEMS;
  float* Yb = (float*)smem;                      // reused after the k-loop

  const int tid = threadIdx.x, lane = tid & 31, wave = tid >> 5;
  const int lrow = lane & 15, lg = lane >> 4;
  const int g0 = blockIdx.x * 4;   // global point row (b*N+n), 4 points per block
  const int b = g0 / NPTS;

  v8f acc[NTPW * 5];
  #pragma unroll
  for (int t = 0; t < NTPW * 5; ++t)
    #pragma unroll
    for (int j = 0; j < 8; ++j) acc[t][j] = 0.f;

  for (int ks = 0; ks < KS; ++ks) {
    __syncthreads();
    if (HAVE_TDM && wave == 0)
      tdm_stage_weights(Wb + ks * 32, Bs, O, CPAD);   // async DMA overlaps gather below
    if constexpr (CIN % 4 == 0) {
      // vector gather: chunks never straddle the (x_j - x_i | x_i) boundary
      for (int s = tid; s < 80 * 8; s += NWAVES * 32) {
        int r = s >> 3, q = s & 7;
        int p = r / 20, j = r % 20;
        int nrow = g0 + p;
        int m = idx[(size_t)nrow * KNN + j];
        int mrow = b * NPTS + m;
        int c0 = ks * 32 + q * 4;
        v4f v;
        if (c0 < CIN) {
          v4f vn = ((const v4f*)X)[(size_t)mrow * (ldX >> 2) + (c0 >> 2)];
          v4f vc = ((const v4f*)X)[(size_t)nrow * (ldX >> 2) + (c0 >> 2)];
          #pragma unroll
          for (int jj = 0; jj < 4; ++jj) v[jj] = vn[jj] - vc[jj];
        } else {
          v = ((const v4f*)X)[(size_t)nrow * (ldX >> 2) + ((c0 - CIN) >> 2)];
        }
        *(v4us*)(As + r * ASTR + q * 4) = pack4(v);
      }
    } else {
      for (int s = tid; s < 80 * 32; s += NWAVES * 32) {
        int r = s >> 5, cc = s & 31;
        int p = r / 20, j = r % 20;
        int nrow = g0 + p;
        int m = idx[(size_t)nrow * KNN + j];
        int mrow = b * NPTS + m;
        int c = ks * 32 + cc;
        float v;
        if (c < CIN)          v = X[(size_t)mrow * ldX + c] - X[(size_t)nrow * ldX + c];
        else if (c < 2 * CIN) v = X[(size_t)nrow * ldX + (c - CIN)];
        else                  v = 0.f;
        As[r * ASTR + cc] = f2bf(v);
      }
    }
#if HAVE_TDM
    if (wave == 0) __builtin_amdgcn_s_wait_tensorcnt(0);
#else
    copy_stage_weights(Wb + ks * 32, Bs, O, CPAD, tid, NWAVES * 32);
#endif
    __syncthreads();
    BF16V a[5];
    #pragma unroll
    for (int mt = 0; mt < 5; ++mt)
      #pragma unroll
      for (int e = 0; e < 16; ++e)
        a[mt].u[e] = As[(mt * 16 + lrow) * ASTR + (lg * 16 + e)];
    #pragma unroll
    for (int t = 0; t < NTPW; ++t) {
      int ntile = wave + NWAVES * t;
      BF16V bf;
      #pragma unroll
      for (int e = 0; e < 16; ++e)
        bf.u[e] = Bs[(ntile * 16 + lrow) * ASTR + (lg * 16 + e)];
      #pragma unroll
      for (int mt = 0; mt < 5; ++mt)
        acc[t * 5 + mt] = __builtin_amdgcn_wmma_f32_16x16x32_bf16(false, a[mt].b, false, bf.b, (short)0, acc[t * 5 + mt], false, false);
    }
  }

  float* Y = Yb + wave * 80 * 16;
  for (int t = 0; t < NTPW; ++t) {
    __syncthreads();
    int ntile = wave + NWAVES * t;
    int o = ntile * 16 + lrow;
    float scale = gam[o] * BNRS;
    float bias = bet[o];
    #pragma unroll
    for (int mt = 0; mt < 5; ++mt)
      #pragma unroll
      for (int v = 0; v < 8; ++v) {
        int r = mt * 16 + lg * 8 + v;
        Y[r * 16 + lrow] = leakyf(acc[t * 5 + mt][v] * scale + bias);
      }
    __syncthreads();
    for (int it = lane; it < 64; it += 32) {
      int p = it >> 4, col = it & 15;
      float mx = -FLT_MAX;
      #pragma unroll
      for (int j = 0; j < KNN; ++j) mx = fmaxf(mx, Y[(p * 20 + j) * 16 + col]);
      out[(size_t)(g0 + p) * ldout + ntile * 16 + col] = mx;
    }
  }
}

// ---------------------------------------------------------------------------
// Generic bf16 GEMM + affine(+leaky) epilogue; optional fused max-over-N pool.
// Y(M x Opad) = act( X(M x Kreal) * W(Opad x Kpad)^T * scale + bias )
// block 256 threads (8 waves); tile 128 rows x 64 cols; weights TDM-staged.
// ---------------------------------------------------------------------------
__global__ void k_gemm_bf16(const float* __restrict__ X, int Mreal, int Kreal, int ksteps,
                            const unsigned short* __restrict__ Wb, int Kpad, int Opad,
                            const float* __restrict__ gam, const float* __restrict__ bbn,
                            const float* __restrict__ blin, int act,
                            float* __restrict__ out, int ldout, int Oreal,
                            float* __restrict__ pool, int poolcols) {
  __shared__ __align__(16) unsigned short As[128 * ASTR];
  __shared__ __align__(16) unsigned short Bs[64 * ASTR];
  const int tid = threadIdx.x, lane = tid & 31, wave = tid >> 5;
  const int lrow = lane & 15, lg = lane >> 4;
  const int rowBase = blockIdx.x * 128, colBase = blockIdx.y * 64;

  v8f acc[4];
  #pragma unroll
  for (int t = 0; t < 4; ++t)
    #pragma unroll
    for (int j = 0; j < 8; ++j) acc[t][j] = 0.f;

  for (int ks = 0; ks < ksteps; ++ks) {
    __syncthreads();
    if (HAVE_TDM && wave == 0)
      tdm_stage_weights(Wb + (size_t)colBase * Kpad + ks * 32, Bs, 64, Kpad);
    for (int s = tid; s < 128 * 8; s += 256) {   // float4 chunks (Kreal % 4 == 0)
      int r = s >> 3, q = s & 7;
      int row = rowBase + r, c0 = ks * 32 + q * 4;
      v4f v;
      if (row < Mreal && c0 < Kreal) v = ((const v4f*)X)[(size_t)row * (Kreal >> 2) + (c0 >> 2)];
      else { v[0] = 0.f; v[1] = 0.f; v[2] = 0.f; v[3] = 0.f; }
      *(v4us*)(As + r * ASTR + q * 4) = pack4(v);
    }
#if HAVE_TDM
    if (wave == 0) __builtin_amdgcn_s_wait_tensorcnt(0);
#else
    copy_stage_weights(Wb + (size_t)colBase * Kpad + ks * 32, Bs, 64, Kpad, tid, 256);
#endif
    __syncthreads();
    BF16V a;
    #pragma unroll
    for (int e = 0; e < 16; ++e)
      a.u[e] = As[(wave * 16 + lrow) * ASTR + (lg * 16 + e)];
    #pragma unroll
    for (int nt = 0; nt < 4; ++nt) {
      BF16V bf;
      #pragma unroll
      for (int e = 0; e < 16; ++e)
        bf.u[e] = Bs[(nt * 16 + lrow) * ASTR + (lg * 16 + e)];
      acc[nt] = __builtin_amdgcn_wmma_f32_16x16x32_bf16(false, a.b, false, bf.b, (short)0, acc[nt], false, false);
    }
  }

  #pragma unroll
  for (int nt = 0; nt < 4; ++nt) {
    int o = colBase + nt * 16 + lrow;
    float scale = 1.f, bias = 0.f;
    if (o < Oreal) {
      scale = gam ? gam[o] * BNRS : 1.f;
      bias = (bbn ? bbn[o] : 0.f) + (blin ? blin[o] * scale : 0.f);
    }
    if (pool) {
      float mx = -FLT_MAX;
      #pragma unroll
      for (int v = 0; v < 8; ++v) {
        float y = acc[nt][v] * scale + bias;
        if (act) y = leakyf(y);
        mx = fmaxf(mx, y);
      }
      int row0 = rowBase + wave * 16 + lg * 8;
      int bb = row0 / NPTS;
      atomicMaxF(&pool[(size_t)bb * poolcols + o], mx);
    } else {
      #pragma unroll
      for (int v = 0; v < 8; ++v) {
        int row = rowBase + wave * 16 + lg * 8 + v;
        float y = acc[nt][v] * scale + bias;
        if (act) y = leakyf(y);
        if (row < Mreal && o < Oreal) out[(size_t)row * ldout + o] = y;
      }
    }
  }
}

// ---------------------------------------------------------------------------
extern "C" void kernel_launch(void* const* d_in, const int* in_sizes, int n_in,
                              void* d_out, int out_size, void* d_ws, size_t ws_size,
                              hipStream_t stream) {
  const float* x   = (const float*)d_in[0];
  const float* W1  = (const float*)d_in[1];  const float* g1  = (const float*)d_in[2];  const float* b1 = (const float*)d_in[3];
  const float* W2  = (const float*)d_in[4];  const float* g2  = (const float*)d_in[5];  const float* b2 = (const float*)d_in[6];
  const float* W2m = (const float*)d_in[7];  const float* g2m = (const float*)d_in[8];  const float* b2m = (const float*)d_in[9];
  const float* W3  = (const float*)d_in[10]; const float* g3  = (const float*)d_in[11]; const float* b3 = (const float*)d_in[12];
  const float* W4  = (const float*)d_in[13]; const float* g4  = (const float*)d_in[14]; const float* b4 = (const float*)d_in[15];
  const float* W5  = (const float*)d_in[16]; const float* g5  = (const float*)d_in[17]; const float* b5 = (const float*)d_in[18];
  const float* L1  = (const float*)d_in[19]; const float* g6  = (const float*)d_in[20]; const float* b6 = (const float*)d_in[21];
  const float* L2  = (const float*)d_in[22]; const float* bL2 = (const float*)d_in[23];
  const float* g7  = (const float*)d_in[24]; const float* b7  = (const float*)d_in[25];
  const float* L3  = (const float*)d_in[26]; const float* bL3 = (const float*)d_in[27];

  char* ws = (char*)d_ws;
  size_t off = 0;
  auto alloc = [&](size_t bytes) -> void* {
    off = (off + 255) & ~(size_t)255;
    void* p = ws + off;
    off += bytes;
    return p;
  };

  const int R = BATCH * NPTS;  // 8192 point-rows
  float* xt     = (float*)alloc((size_t)R * 4 * 4);
  float* xcat12 = (float*)alloc((size_t)R * 128 * 4);
  float* nf1    = (float*)alloc((size_t)R * 256 * 4);
  float* xcat34 = (float*)alloc((size_t)R * 512 * 4);
  float* G      = (float*)alloc((size_t)BATCH * NPTS * NPTS * 4);
  float* xx     = (float*)alloc((size_t)R * 4);
  int*   idx    = (int*)alloc((size_t)R * KNN * 4);
  float* p1     = (float*)alloc((size_t)BATCH * 1024 * 4);
  float* p2     = (float*)alloc((size_t)BATCH * 1024 * 4);
  float* f0     = (float*)alloc((size_t)16 * 2048 * 4);
  float* f1     = (float*)alloc((size_t)16 * 512 * 4);
  float* f2     = (float*)alloc((size_t)16 * 256 * 4);
  unsigned short* Wb1  = (unsigned short*)alloc((size_t)64 * 32 * 2);
  unsigned short* Wb2  = (unsigned short*)alloc((size_t)64 * 128 * 2);
  unsigned short* Wb3  = (unsigned short*)alloc((size_t)256 * 512 * 2);
  unsigned short* Wb4  = (unsigned short*)alloc((size_t)256 * 512 * 2);
  unsigned short* Wb2m = (unsigned short*)alloc((size_t)1024 * 128 * 2);
  unsigned short* Wb5  = (unsigned short*)alloc((size_t)1024 * 512 * 2);
  unsigned short* WbL1 = (unsigned short*)alloc((size_t)512 * 2048 * 2);
  unsigned short* WbL2 = (unsigned short*)alloc((size_t)256 * 512 * 2);
  unsigned short* WbL3 = (unsigned short*)alloc((size_t)64 * 256 * 2);

  auto cvg = [](int n) { return (n + 255) / 256; };

  // weight conversion (fp32 -> bf16, original OxK orientation, zero-padded)
  k_convert_w<<<cvg(64 * 32), 256, 0, stream>>>(W1, Wb1, 64, 6, 64, 32);
  k_convert_w<<<cvg(64 * 128), 256, 0, stream>>>(W2, Wb2, 64, 128, 64, 128);
  k_convert_w<<<cvg(256 * 512), 256, 0, stream>>>(W3, Wb3, 256, 512, 256, 512);
  k_convert_w<<<cvg(256 * 512), 256, 0, stream>>>(W4, Wb4, 256, 512, 256, 512);
  k_convert_w<<<cvg(1024 * 128), 256, 0, stream>>>(W2m, Wb2m, 1024, 128, 1024, 128);
  k_convert_w<<<cvg(1024 * 512), 256, 0, stream>>>(W5, Wb5, 1024, 512, 1024, 512);
  k_convert_w<<<cvg(512 * 2048), 256, 0, stream>>>(L1, WbL1, 512, 2048, 512, 2048);
  k_convert_w<<<cvg(256 * 512), 256, 0, stream>>>(L2, WbL2, 256, 512, 256, 512);
  k_convert_w<<<cvg(64 * 256), 256, 0, stream>>>(L3, WbL3, 40, 256, 64, 256);

  // EdgeConv1 (knn on xyz)
  k_transpose_x<<<cvg(R), 256, 0, stream>>>(x, xt);
  k_rownorm<<<cvg(R), 256, 0, stream>>>(xt, 4, 3, xx, R);
  k_knn_xyz<<<cvg(R), 256, 0, stream>>>(xt, xx, idx);
  k_edgeconv<3, 32, 64><<<R / 4, 128, 0, stream>>>(xt, 4, idx, Wb1, g1, b1, xcat12, 128);

  // EdgeConv2 (knn on x1 = xcat12[:, :64])
  k_rownorm<<<cvg(R), 256, 0, stream>>>(xcat12, 128, 64, xx, R);
  k_gram_split<<<dim3(8, 8, BATCH), 256, 0, stream>>>(xcat12, 128, 2, G);
  k_topk_gram<<<cvg(R), 256, 0, stream>>>(G, xx, idx);
  k_edgeconv<64, 128, 64><<<R / 4, 128, 0, stream>>>(xcat12, 128, idx, Wb2, g2, b2, xcat12 + 64, 128);

  // EdgeConv3 (knn on nf1 = cat(xcat12, xcat12))
  k_build_nf1<<<cvg(R * 64), 256, 0, stream>>>(xcat12, nf1);
  k_rownorm<<<cvg(R), 256, 0, stream>>>(nf1, 256, 256, xx, R);
  k_gram_split<<<dim3(8, 8, BATCH), 256, 0, stream>>>(nf1, 256, 8, G);
  k_topk_gram<<<cvg(R), 256, 0, stream>>>(G, xx, idx);
  k_edgeconv<256, 512, 256><<<R / 4, 256, 0, stream>>>(nf1, 256, idx, Wb3, g3, b3, xcat34, 512);

  // EdgeConv4 (knn on x3 = xcat34[:, :256])
  k_rownorm<<<cvg(R), 256, 0, stream>>>(xcat34, 512, 256, xx, R);
  k_gram_split<<<dim3(8, 8, BATCH), 256, 0, stream>>>(xcat34, 512, 8, G);
  k_topk_gram<<<cvg(R), 256, 0, stream>>>(G, xx, idx);
  k_edgeconv<256, 512, 256><<<R / 4, 256, 0, stream>>>(xcat34, 512, idx, Wb4, g4, b4, xcat34 + 256, 512);

  // projection GEMMs with fused max-over-N pooling (x_t1/x_t2 never materialized:
  // saves 64MB of traffic; whole working set stays L2-resident)
  k_fill<<<cvg(BATCH * 1024), 256, 0, stream>>>(p1, BATCH * 1024, -FLT_MAX);
  k_fill<<<cvg(BATCH * 1024), 256, 0, stream>>>(p2, BATCH * 1024, -FLT_MAX);
  k_gemm_bf16<<<dim3(64, 16), 256, 0, stream>>>(xcat12, R, 128, 4, Wb2m, 128, 1024,
                                                g2m, b2m, nullptr, 1,
                                                nullptr, 0, 1024, p1, 1024);
  k_gemm_bf16<<<dim3(64, 16), 256, 0, stream>>>(xcat34, R, 512, 16, Wb5, 512, 1024,
                                                g5, b5, nullptr, 1,
                                                nullptr, 0, 1024, p2, 1024);

  // classifier head
  k_build_f0<<<cvg(16 * 2048), 256, 0, stream>>>(p1, p2, f0);
  k_gemm_bf16<<<dim3(1, 8), 256, 0, stream>>>(f0, 8, 2048, 64, WbL1, 2048, 512,
                                              g6, b6, nullptr, 1, f1, 512, 512, nullptr, 0);
  k_gemm_bf16<<<dim3(1, 4), 256, 0, stream>>>(f1, 8, 512, 16, WbL2, 512, 256,
                                              g7, b7, bL2, 1, f2, 256, 256, nullptr, 0);
  k_gemm_bf16<<<dim3(1, 1), 256, 0, stream>>>(f2, 8, 256, 8, WbL3, 256, 64,
                                              nullptr, nullptr, bL3, 0,
                                              (float*)d_out, 40, 40, nullptr, 0);
}
